// TemporalMMDLoss_88003879895559
// MI455X (gfx1250) — matrix operand order repaired and checked by reading, not compiled, not run on hardware
//
#include <hip/hip_runtime.h>

typedef __attribute__((ext_vector_type(16))) __bf16 v16bf;
typedef __attribute__((ext_vector_type(8)))  float  v8f;

#define D        512
#define NREG     4
#define MAXN     8192
#define MAXROWS  (2 * MAXN)
#define TILE     64
#define LDSPADB  72   // bf16 elements per LDS row (144B, 16B-aligned)

// workspace layout (byte offsets)
#define OFF_META  0                               // int[16]  : per r {b, rowoff, cnt_s, cnt_t}
#define OFF_ACC   128                             // float[12]: per r {xx, yy, xy}
#define OFF_BW    256                             // float[4] : bandwidth per regime
#define OFF_SIDX  1024                            // int[4*MAXN]
#define OFF_TIDX  (OFF_SIDX + 4 * MAXN * 4)       // int[4*MAXN]
#define OFF_SQ    (OFF_TIDX + 4 * MAXN * 4)       // float[MAXROWS]
#define OFF_PACKH (OFF_SQ + MAXROWS * 4 + 256)    // ushort[MAXROWS*D] bf16 hi plane
#define OFF_PACKL (OFF_PACKH + MAXROWS * D * 2)   // ushort[MAXROWS*D] bf16 lo plane

static __device__ __forceinline__ unsigned short f2bf(float x) {
  unsigned u = __float_as_uint(x);
  u += 0x7FFFu + ((u >> 16) & 1u);  // round to nearest even
  return (unsigned short)(u >> 16);
}
static __device__ __forceinline__ float bf2f(unsigned short h) {
  return __uint_as_float(((unsigned)h) << 16);
}

// ---------------------------------------------------------------------------
// Kernel 1: ordered compaction per (regime, side) via wave32 ballot,
// then metadata (b, row offsets) and accumulator zeroing.
// ---------------------------------------------------------------------------
__global__ __launch_bounds__(256) void compact_kernel(
    const long long* __restrict__ sreg, const long long* __restrict__ treg,
    int Ns, int Nt, char* __restrict__ ws) {
  int* meta   = (int*)(ws + OFF_META);
  float* acc  = (float*)(ws + OFF_ACC);
  int* sidx   = (int*)(ws + OFF_SIDX);
  int* tidx   = (int*)(ws + OFF_TIDX);
  __shared__ int scnt[8];

  int tid = threadIdx.x, w = tid >> 5, lane = tid & 31;
  int r = w & 3, side = w >> 2;
  const long long* reg = side ? treg : sreg;
  int N = side ? Nt : Ns;
  if (N > MAXN) N = MAXN;
  int* out = (side ? tidx : sidx) + r * MAXN;

  int cnt = 0;
  for (int base = 0; base < N; base += 32) {
    int i = base + lane;
    bool m = (i < N) && ((int)reg[i] == r);
    unsigned mask = (unsigned)__ballot(m);
    if (m) {
      int pre = __popc(mask & ((1u << lane) - 1));
      out[cnt + pre] = i;
    }
    cnt += __popc(mask);
  }
  if (lane == 0) scnt[w] = cnt;
  __syncthreads();

  if (tid == 0) {
    int off = 0;
    for (int rr = 0; rr < NREG; ++rr) {
      int cs = scnt[rr], ct = scnt[4 + rr];
      int bb = cs < ct ? cs : ct;
      meta[rr * 4 + 0] = bb;
      meta[rr * 4 + 1] = off;
      meta[rr * 4 + 2] = cs;
      meta[rr * 4 + 3] = ct;
      off += 2 * bb;
    }
  }
  if (tid < 12) acc[tid] = 0.0f;
}

// ---------------------------------------------------------------------------
// Kernel 2: gather selected rows, split fp32 -> bf16 hi + bf16 lo planes,
// compute exact fp32 per-row squared norms. One block per (row, regime).
// ---------------------------------------------------------------------------
__global__ __launch_bounds__(128) void gather_kernel(
    const float* __restrict__ sfeat, const float* __restrict__ tfeat,
    char* __restrict__ ws) {
  const int* meta = (const int*)(ws + OFF_META);
  const int* sidx = (const int*)(ws + OFF_SIDX);
  const int* tidx = (const int*)(ws + OFF_TIDX);
  float* sq       = (float*)(ws + OFF_SQ);
  unsigned short* packH = (unsigned short*)(ws + OFF_PACKH);
  unsigned short* packL = (unsigned short*)(ws + OFF_PACKL);

  int r = blockIdx.y;
  int b = meta[r * 4 + 0];
  int row = blockIdx.x;
  if (row >= 2 * b) return;
  int off = meta[r * 4 + 1];

  const float* src;
  if (row < b) src = sfeat + (size_t)sidx[r * MAXN + row] * D;
  else         src = tfeat + (size_t)tidx[r * MAXN + row - b] * D;

  int tid = threadIdx.x;  // 128 threads, 4 consecutive floats each
  float4 v = ((const float4*)src)[tid];
  float xv[4] = {v.x, v.y, v.z, v.w};

  union { unsigned short h[4]; uint2 u; } ph, pl;
  float ss = 0.f;
#pragma unroll
  for (int e = 0; e < 4; ++e) {
    float x = xv[e];
    ss += x * x;
    unsigned short hi = f2bf(x);
    ph.h[e] = hi;
    pl.h[e] = f2bf(x - bf2f(hi));
  }
  size_t rowbase = (size_t)(off + row) * D;
  ((uint2*)(packH + rowbase))[tid] = ph.u;
  ((uint2*)(packL + rowbase))[tid] = pl.u;

  for (int s = 16; s > 0; s >>= 1) ss += __shfl_down(ss, s, 32);
  __shared__ float red[4];
  int w = tid >> 5, lane = tid & 31;
  if (lane == 0) red[w] = ss;
  __syncthreads();
  if (tid == 0) sq[off + row] = red[0] + red[1] + red[2] + red[3];
}

// ---------------------------------------------------------------------------
// Kernel 3: closed-form bandwidth per regime:
//   sum(L2) = 2n*S - 2*||sum_i x_i||^2,  S = sum_i sq[i]
//   bw = sum(L2)/(n^2-n) / KERNEL_MUL^(KERNEL_NUM//2)   (= /4)
// ---------------------------------------------------------------------------
__global__ __launch_bounds__(256) void bw_kernel(char* __restrict__ ws) {
  const int* meta = (const int*)(ws + OFF_META);
  const float* sq = (const float*)(ws + OFF_SQ);
  const unsigned short* packH = (const unsigned short*)(ws + OFF_PACKH);
  const unsigned short* packL = (const unsigned short*)(ws + OFF_PACKL);
  float* bw = (float*)(ws + OFF_BW);

  int r = blockIdx.x;
  int b = meta[r * 4 + 0], off = meta[r * 4 + 1];
  int n = 2 * b;
  int tid = threadIdx.x;  // 256 threads, 2 columns each
  if (n < 2) { if (tid == 0) bw[r] = 1.0f; return; }

  const unsigned short* bH = packH + (size_t)off * D;
  const unsigned short* bL = packL + (size_t)off * D;
  float cs0 = 0.f, cs1 = 0.f;
  for (int row = 0; row < n; ++row) {
    size_t rb = (size_t)row * D;
    cs0 += bf2f(bH[rb + tid])       + bf2f(bL[rb + tid]);
    cs1 += bf2f(bH[rb + tid + 256]) + bf2f(bL[rb + tid + 256]);
  }
  float vv = cs0 * cs0 + cs1 * cs1;
  float S = 0.f;
  for (int rr = tid; rr < n; rr += 256) S += sq[off + rr];

  for (int s = 16; s > 0; s >>= 1) {
    S  += __shfl_down(S, s, 32);
    vv += __shfl_down(vv, s, 32);
  }
  __shared__ float rs[8], rv[8];
  int w = tid >> 5, lane = tid & 31;
  if (lane == 0) { rs[w] = S; rv[w] = vv; }
  __syncthreads();
  if (tid == 0) {
    double Sd = 0.0, Vd = 0.0;
    for (int i = 0; i < 8; ++i) { Sd += rs[i]; Vd += rv[i]; }
    double nn = (double)n;
    double sumL2 = 2.0 * nn * Sd - 2.0 * Vd;
    bw[r] = (float)(sumL2 / (nn * nn - nn) / 4.0);
  }
}

// ---------------------------------------------------------------------------
// Kernel 4: fused Gram via split-bf16 WMMA (hi*hi + hi*lo + lo*hi) +
// multi-scale exp + block-classified reduction. 64x64 tile per block,
// 4 waves; each wave: 16-row slab -> 4 C tiles (v_wmma_f32_16x16x32_bf16).
// Lower triangle only (weight 2 off-diagonal).
// ---------------------------------------------------------------------------
__global__ __launch_bounds__(128) void gram_kernel(char* __restrict__ ws) {
  const int* meta = (const int*)(ws + OFF_META);
  const float* sq = (const float*)(ws + OFF_SQ);
  const float* bw = (const float*)(ws + OFF_BW);
  float* acc      = (float*)(ws + OFF_ACC);
  const unsigned short* packH = (const unsigned short*)(ws + OFF_PACKH);
  const unsigned short* packL = (const unsigned short*)(ws + OFF_PACKL);

  int r = blockIdx.z;
  int b = meta[r * 4 + 0], off = meta[r * 4 + 1];
  int n = 2 * b;
  if (b < 2) return;
  int it = blockIdx.x, jt = blockIdx.y;
  if (jt > it) return;
  if (it * TILE >= n) return;
  float wgt = (jt < it) ? 2.0f : 1.0f;

  __shared__ unsigned short AtH[TILE][LDSPADB];
  __shared__ unsigned short AtL[TILE][LDSPADB];
  __shared__ unsigned short BtH[TILE][LDSPADB];
  __shared__ unsigned short BtL[TILE][LDSPADB];

  int tid = threadIdx.x, w = tid >> 5, lane = tid & 31;
  int i0 = it * TILE, j0 = jt * TILE;
  const unsigned short* baseH = packH + (size_t)off * D;
  const unsigned short* baseL = packL + (size_t)off * D;

  v8f c[4];
#pragma unroll
  for (int jj = 0; jj < 4; ++jj)
#pragma unroll
    for (int e = 0; e < 8; ++e) c[jj][e] = 0.0f;

  int lm = lane & 15;
  int sel = lane >> 4;           // lane half selects K sub-range
  const int aK = sel * 8;        // A: low lanes K0-7/K16-23, high lanes K8-15/K24-31
  const int bK = sel * 8;        // B: low lanes K0-7/K16-23, high lanes K8-15/K24-31

  union FragBF { v16bf v; uint4 u[2]; };

  for (int kc = 0; kc < D; kc += TILE) {
    // stage A (rows i0..) and B (rows j0..) hi/lo k-chunks into LDS
    for (int v = tid; v < TILE * 8; v += 128) {
      int row = v >> 3, cc = (v & 7) << 3;  // 8 bf16 (16B) per slot
      uint4 z = {0u, 0u, 0u, 0u};
      int gi = i0 + row, gj = j0 + row;
      size_t gio = (size_t)gi * D + kc + cc;
      size_t gjo = (size_t)gj * D + kc + cc;
      *(uint4*)&AtH[row][cc] = (gi < n) ? *(const uint4*)(baseH + gio) : z;
      *(uint4*)&AtL[row][cc] = (gi < n) ? *(const uint4*)(baseL + gio) : z;
      *(uint4*)&BtH[row][cc] = (gj < n) ? *(const uint4*)(baseH + gjo) : z;
      *(uint4*)&BtL[row][cc] = (gj < n) ? *(const uint4*)(baseL + gjo) : z;
    }
    __syncthreads();

    int arow = (w << 4) + lm;
#pragma unroll
    for (int ks = 0; ks < TILE; ks += 32) {
      FragBF ah, al;
      ah.u[0] = *(const uint4*)&AtH[arow][ks + aK];
      ah.u[1] = *(const uint4*)&AtH[arow][ks + 16 + aK];
      al.u[0] = *(const uint4*)&AtL[arow][ks + aK];
      al.u[1] = *(const uint4*)&AtL[arow][ks + 16 + aK];
#pragma unroll
      for (int jj = 0; jj < 4; ++jj) {
        int brow = (jj << 4) + lm;
        FragBF bh, bl;
        bh.u[0] = *(const uint4*)&BtH[brow][ks + bK];
        bh.u[1] = *(const uint4*)&BtH[brow][ks + bK + 16];
        bl.u[0] = *(const uint4*)&BtL[brow][ks + bK];
        bl.u[1] = *(const uint4*)&BtL[brow][ks + bK + 16];
        c[jj] = __builtin_amdgcn_wmma_f32_16x16x32_bf16(
            false, ah.v, false, bh.v, (short)0, c[jj], false, false);
        c[jj] = __builtin_amdgcn_wmma_f32_16x16x32_bf16(
            false, ah.v, false, bl.v, (short)0, c[jj], false, false);
        c[jj] = __builtin_amdgcn_wmma_f32_16x16x32_bf16(
            false, al.v, false, bh.v, (short)0, c[jj], false, false);
      }
    }
    __syncthreads();
  }

  // epilogue: L2 -> multi-scale Gaussian kernel -> XX/YY/XY accumulation
  float bwv = bw[r];
  float inv[5];
  float scale = 1.0f;
#pragma unroll
  for (int m = 0; m < 5; ++m) {
    inv[m] = -1.0f / (bwv * scale + 1e-8f);
    scale *= 2.0f;
  }

  float xx = 0.f, yy = 0.f, xy = 0.f;
  int mhi = sel << 3;
#pragma unroll
  for (int jj = 0; jj < 4; ++jj) {
    int gj = j0 + (jj << 4) + lm;
    if (gj >= n) continue;
    float sqj = sq[off + gj];
#pragma unroll
    for (int v = 0; v < 8; ++v) {
      int gi = i0 + (w << 4) + mhi + v;
      if (gi >= n) continue;
      float l2 = sq[off + gi] + sqj - 2.0f * c[jj][v];
      float kv = 0.f;
#pragma unroll
      for (int m = 0; m < 5; ++m) kv += __expf(l2 * inv[m]);
      kv *= 0.2f;
      bool iS = gi < b, jS = gj < b;
      if (iS && jS)        xx += kv;
      else if (!iS && !jS) yy += kv;
      else                 xy += kv;
    }
  }
  xx *= wgt; yy *= wgt; xy *= wgt;
  for (int s = 16; s > 0; s >>= 1) {
    xx += __shfl_down(xx, s, 32);
    yy += __shfl_down(yy, s, 32);
    xy += __shfl_down(xy, s, 32);
  }
  if (lane == 0) {
    atomicAdd(&acc[r * 3 + 0], xx);
    atomicAdd(&acc[r * 3 + 1], yy);
    atomicAdd(&acc[r * 3 + 2], xy);
  }
}

// ---------------------------------------------------------------------------
// Kernel 5: finalize. loss_r = (XXsum + YYsum - XYsum) / b^2, weighted 1/4.
// ---------------------------------------------------------------------------
__global__ void final_kernel(const char* __restrict__ ws, float* __restrict__ out) {
  if (threadIdx.x != 0 || blockIdx.x != 0) return;
  const int* meta  = (const int*)(ws + OFF_META);
  const float* acc = (const float*)(ws + OFF_ACC);
  float loss = 0.0f;
  for (int r = 0; r < NREG; ++r) {
    int b = meta[r * 4 + 0], cs = meta[r * 4 + 2], ct = meta[r * 4 + 3];
    if (cs >= 2 && ct >= 2 && b >= 2) {
      float invb2 = 1.0f / ((float)b * (float)b);
      loss += 0.25f * (acc[r * 3 + 0] + acc[r * 3 + 1] - acc[r * 3 + 2]) * invb2;
    }
  }
  out[0] = loss;
}

extern "C" void kernel_launch(void* const* d_in, const int* in_sizes, int n_in,
                              void* d_out, int out_size, void* d_ws, size_t ws_size,
                              hipStream_t stream) {
  const float* sfeat    = (const float*)d_in[0];
  const float* tfeat    = (const float*)d_in[1];
  const long long* sreg = (const long long*)d_in[2];
  const long long* treg = (const long long*)d_in[3];
  float* out = (float*)d_out;
  char* ws   = (char*)d_ws;
  int Ns = in_sizes[2];
  int Nt = in_sizes[3];

  compact_kernel<<<dim3(1), dim3(256), 0, stream>>>(sreg, treg, Ns, Nt, ws);
  gather_kernel<<<dim3(MAXROWS, NREG), dim3(128), 0, stream>>>(sfeat, tfeat, ws);
  bw_kernel<<<dim3(NREG), dim3(256), 0, stream>>>(ws);
  gram_kernel<<<dim3(MAXROWS / TILE, MAXROWS / TILE, NREG), dim3(128), 0, stream>>>(ws);
  final_kernel<<<1, 32, 0, stream>>>(ws, out);
}
